// PoseGRU_72524817760644
// MI455X (gfx1250) — compile-verified, matrix operand order checked by reading, not run on hardware
//
#include <hip/hip_runtime.h>

typedef __attribute__((ext_vector_type(16))) __bf16 v16bf;
typedef __attribute__((ext_vector_type(8)))  float  v8f;

#define NJ      24
#define HDIM    32
#define IDIM    448
#define KTOT    480
#define KSTEPS  15          // 480 / 32
#define ROWS    64          // rows per block (4 waves x 16 rows)
#define NTHREADS 128
#define NWAVES  4

// Codegen barrier: memory ops cannot be reordered across this.
#define SCHED_FENCE() asm volatile("" ::: "memory")

// Parents for joints 1..23, 5 bits each, packed into two 64-bit immediates
// (joint 0 is the root and handled specially). Pure SALU decode, no s_load.
static constexpr unsigned long long PLO =   // joints 1..12
    (0ULL<<0)|(0ULL<<5)|(0ULL<<10)|(1ULL<<15)|(2ULL<<20)|(3ULL<<25)|
    (4ULL<<30)|(5ULL<<35)|(6ULL<<40)|(7ULL<<45)|(8ULL<<50)|(9ULL<<55);
static constexpr unsigned long long PHI =   // joints 13..23
    (9ULL<<0)|(9ULL<<5)|(12ULL<<10)|(13ULL<<15)|(14ULL<<20)|(16ULL<<25)|
    (17ULL<<30)|(18ULL<<35)|(19ULL<<40)|(20ULL<<45)|(21ULL<<50);

// ---------------------------------------------------------------------------
// Prep: fp32 weights -> bf16, swizzled into WMMA B-fragment layout.
// Fragment id f = ((j*3+g)*15+kk)*2+nt ; lane holds column n = nt*16+lane%16,
// K = kk*32 + 16*(lane/16) + t; each lane's 16 halves contiguous (32B).
// ---------------------------------------------------------------------------
__global__ __launch_bounds__(256) void prep_weights_kernel(
    const float* __restrict__ Wz, const float* __restrict__ Wr,
    const float* __restrict__ Wq, __bf16* __restrict__ wsw)
{
  int idx = blockIdx.x * 256 + threadIdx.x;
  const int TOTAL = NJ * 3 * KSTEPS * 2 * 32 * 16;
  if (idx >= TOTAL) return;
  int t    = idx & 15;
  int lane = (idx >> 4) & 31;
  int f    = idx >> 9;
  int nt   = f & 1;
  int f2   = f >> 1;
  int kk   = f2 % KSTEPS;
  int f3   = f2 / KSTEPS;
  int g    = f3 % 3;
  int j    = f3 / 3;
  int K = kk * 32 + ((lane >> 4) << 4) + t;
  int n = nt * 16 + (lane & 15);
  const float* W = (g == 0) ? Wz : ((g == 1) ? Wr : Wq);
  wsw[idx] = (__bf16)W[(j * KTOT + K) * HDIM + n];
}

// Raw-hardware transcendentals: v_exp_f32 (base-2) + v_rcp_f32.
__device__ __forceinline__ float fast_sigmoid(float v) {
  float e = __builtin_amdgcn_exp2f(v * -1.442695041f);      // exp(-v)
  return __builtin_amdgcn_rcpf(1.0f + e);
}
__device__ __forceinline__ float fast_tanh(float v) {
  float e = __builtin_amdgcn_exp2f(v * -2.885390082f);      // exp(-2v)
  return 2.0f * __builtin_amdgcn_rcpf(1.0f + e) - 1.0f;
}

__device__ __forceinline__ v16bf load_wfrag(const __bf16* __restrict__ wsw,
                                            int j, int g, int kk, int nt, int lane) {
  const __bf16* p = wsw +
      ((size_t)((((j * 3 + g) * KSTEPS + kk) * 2 + nt) * 32 + lane) << 4);
  return *(const v16bf*)p;
}

#define WMMA_BF16(a, b, c) __builtin_amdgcn_wmma_f32_16x16x32_bf16( \
    false, (a), false, (b), (short)0, (c), false, false)

// ---------------------------------------------------------------------------
// Fused tree-GRU. 4 waves/block, 16 rows/wave. Weight B-fragments flow
// through a software-pipelined 6-fragment ring carried ACROSS joints, with
// explicit codegen fences so loads for step k+1 are issued before the WMMAs
// of step k (no WMMA waits on the load it just issued).
// ---------------------------------------------------------------------------
__global__ __launch_bounds__(NTHREADS) void pose_gru_kernel(
    const float* __restrict__ hpose, const float* __restrict__ xin,
    const __bf16* __restrict__ wsw,
    const float* __restrict__ bz, const float* __restrict__ br,
    const float* __restrict__ bq, float* __restrict__ out)
{
  __shared__ __align__(32) __bf16 xs[ROWS][IDIM];          // 57,344 B
  __shared__ __align__(32) __bf16 scr[NWAVES][16][HDIM];   //  4,096 B bounce

  const int tid  = threadIdx.x;
  const int wave = tid >> 5;
  const int lane = tid & 31;
  const int rowBase = blockIdx.x * ROWS;

  // Stage x tile -> LDS bf16 (coalesced float4 loads)
  for (int e = tid * 4; e < ROWS * IDIM; e += NTHREADS * 4) {
    int r = e / IDIM;
    int c = e - r * IDIM;
    float4 v = *(const float4*)(xin + (size_t)(rowBase + r) * IDIM + c);
    xs[r][c + 0] = (__bf16)v.x;
    xs[r][c + 1] = (__bf16)v.y;
    xs[r][c + 2] = (__bf16)v.z;
    xs[r][c + 3] = (__bf16)v.w;
  }
  __syncthreads();

  const int m    = lane & 15;     // row within the 16-row M tile
  const int kh   = lane >> 4;     // 16-wide K half held by this lane
  const int lrow = wave * 16 + m; // row within block (xs index)
  const int waveRow = rowBase + wave * 16;

  // Prime the cross-joint B-fragment ring with joint 0, k-step 1.
  v16bf cur0 = load_wfrag(wsw, 0, 0, 1, 0, lane);
  v16bf cur1 = load_wfrag(wsw, 0, 0, 1, 1, lane);
  v16bf cur2 = load_wfrag(wsw, 0, 1, 1, 0, lane);
  v16bf cur3 = load_wfrag(wsw, 0, 1, 1, 1, lane);
  v16bf cur4 = load_wfrag(wsw, 0, 2, 1, 0, lane);
  v16bf cur5 = load_wfrag(wsw, 0, 2, 1, 1, lane);
  SCHED_FENCE();

  #pragma unroll 1
  for (int i = 0; i < NJ; ++i) {
    // Uniform SALU parent decode (no scalar memory on the critical path)
    const int e = i - 1;
    const int p = (int)(((e < 12) ? (PLO >> (5 * e)) : (PHI >> (5 * (e - 12)))) & 31ULL);

    // Bias (issued early; consumed only in the tail)
    float bzv[2], brv[2], bqv[2];
    #pragma unroll
    for (int nt = 0; nt < 2; ++nt) {
      bzv[nt] = bz[i * HDIM + nt * 16 + m];
      brv[nt] = br[i * HDIM + nt * 16 + m];
      bqv[nt] = bq[i * HDIM + nt * 16 + m];
    }

    // Parent-h A-fragment (row m, cols kh*16..+15); stage into bounce LDS
    const float* hsrc = (i == 0)
        ? (hpose + (size_t)(waveRow + m) * (NJ * HDIM))             // hpose[:,0,:]
        : (out   + (size_t)(waveRow + m) * (NJ * HDIM) + p * HDIM); // updated parent
    const float4* hp4 = (const float4*)(hsrc + kh * 16);
    float hv[16];
    *(float4*)&hv[0]  = hp4[0];
    *(float4*)&hv[4]  = hp4[1];
    *(float4*)&hv[8]  = hp4[2];
    *(float4*)&hv[12] = hp4[3];
    v16bf hfrag;
    #pragma unroll
    for (int t = 0; t < 16; ++t) hfrag[t] = (__bf16)hv[t];
    *(v16bf*)&scr[wave][m][kh * 16] = hfrag;

    // Zero-init accumulators (first WMMA in each chain takes inline SRC2=0)
    v8f zacc[2], racc[2], qacc[2];
    {
      v8f zz = {};
      zacc[0] = zz; zacc[1] = zz;
      racc[0] = zz; racc[1] = zz;
      qacc[0] = zz; qacc[1] = zz;
    }

    // x-part k-steps 1..14, one-step-ahead B prefetch enforced by fences.
    // Last iteration prefetches k-step 0 (z0,z0,r0,r0,q0,q0) for the epilogue.
    #pragma unroll
    for (int kk = 1; kk < KSTEPS; ++kk) {
      const int nk = (kk < KSTEPS - 1) ? (kk + 1) : 0;
      v16bf n0 = load_wfrag(wsw, i, 0, nk, 0, lane);
      v16bf n1 = load_wfrag(wsw, i, 0, nk, 1, lane);
      v16bf n2 = load_wfrag(wsw, i, 1, nk, 0, lane);
      v16bf n3 = load_wfrag(wsw, i, 1, nk, 1, lane);
      v16bf n4 = load_wfrag(wsw, i, 2, nk, 0, lane);
      v16bf n5 = load_wfrag(wsw, i, 2, nk, 1, lane);
      SCHED_FENCE();   // loads for k+1 must be issued before WMMAs of step k

      v16bf a = *(const v16bf*)&xs[lrow][(kk - 1) * 32 + kh * 16];
      zacc[0] = WMMA_BF16(a, cur0, zacc[0]);
      zacc[1] = WMMA_BF16(a, cur1, zacc[1]);
      racc[0] = WMMA_BF16(a, cur2, racc[0]);
      racc[1] = WMMA_BF16(a, cur3, racc[1]);
      qacc[0] = WMMA_BF16(a, cur4, qacc[0]);
      qacc[1] = WMMA_BF16(a, cur5, qacc[1]);

      cur0 = n0; cur1 = n1; cur2 = n2; cur3 = n3; cur4 = n4; cur5 = n5;
    }

    // h-dependent k-step 0 for z and r (fragments already in registers)
    zacc[0] = WMMA_BF16(hfrag, cur0, zacc[0]);
    zacc[1] = WMMA_BF16(hfrag, cur1, zacc[1]);
    racc[0] = WMMA_BF16(hfrag, cur2, racc[0]);
    racc[1] = WMMA_BF16(hfrag, cur3, racc[1]);

    // Tail: C-layout h, gates, r*h bounce, q k-step 0, combine, store.
    float hc[2][8];
    #pragma unroll
    for (int nt = 0; nt < 2; ++nt)
      #pragma unroll
      for (int v = 0; v < 8; ++v)
        hc[nt][v] = (float)scr[wave][v + 8 * kh][nt * 16 + m];

    float zval[2][8];
    #pragma unroll
    for (int nt = 0; nt < 2; ++nt)
      #pragma unroll
      for (int v = 0; v < 8; ++v) {
        zval[nt][v] = fast_sigmoid(zacc[nt][v] + bzv[nt]);
        float rv   = fast_sigmoid(racc[nt][v] + brv[nt]);
        scr[wave][v + 8 * kh][nt * 16 + m] = (__bf16)(rv * hc[nt][v]);
      }

    v16bf rha = *(const v16bf*)&scr[wave][m][kh * 16];
    qacc[0] = WMMA_BF16(rha, cur4, qacc[0]);
    qacc[1] = WMMA_BF16(rha, cur5, qacc[1]);

    // Cross-joint prefetch: next joint's k-step-1 fragments, issued here so
    // the combine/store tail and next h-load cover their latency.
    {
      const int jn = (i + 1 < NJ) ? (i + 1) : 0;
      cur0 = load_wfrag(wsw, jn, 0, 1, 0, lane);
      cur1 = load_wfrag(wsw, jn, 0, 1, 1, lane);
      cur2 = load_wfrag(wsw, jn, 1, 1, 0, lane);
      cur3 = load_wfrag(wsw, jn, 1, 1, 1, lane);
      cur4 = load_wfrag(wsw, jn, 2, 1, 0, lane);
      cur5 = load_wfrag(wsw, jn, 2, 1, 1, lane);
    }
    SCHED_FENCE();   // keep the prefetch above the store tail

    #pragma unroll
    for (int nt = 0; nt < 2; ++nt)
      #pragma unroll
      for (int v = 0; v < 8; ++v) {
        float q  = fast_tanh(qacc[nt][v] + bqv[nt]);
        float z  = zval[nt][v];
        float hn = (1.0f - z) * hc[nt][v] + z * q;
        out[(size_t)(waveRow + v + 8 * kh) * (NJ * HDIM)
            + i * HDIM + nt * 16 + m] = hn;
      }
  }
}

// ---------------------------------------------------------------------------
extern "C" void kernel_launch(void* const* d_in, const int* in_sizes, int n_in,
                              void* d_out, int out_size, void* d_ws, size_t ws_size,
                              hipStream_t stream) {
  (void)n_in; (void)out_size; (void)ws_size;
  const float* hpose = (const float*)d_in[0];
  const float* x     = (const float*)d_in[1];
  const float* Wz    = (const float*)d_in[2];
  const float* bz    = (const float*)d_in[3];
  const float* Wr    = (const float*)d_in[4];
  const float* br    = (const float*)d_in[5];
  const float* Wq    = (const float*)d_in[6];
  const float* bq    = (const float*)d_in[7];
  float*  out = (float*)d_out;
  __bf16* wsw = (__bf16*)d_ws;   // 2,211,840 B swizzled bf16 weights

  const int TOTALW = NJ * 3 * KSTEPS * 2 * 32 * 16;
  prep_weights_kernel<<<(TOTALW + 255) / 256, 256, 0, stream>>>(Wz, Wr, Wq, wsw);

  const int B = in_sizes[1] / IDIM;     // 32768
  pose_gru_kernel<<<B / ROWS, NTHREADS, 0, stream>>>(hpose, x, wsw, bz, br, bq, out);
}
